// DistNN_88794153877509
// MI455X (gfx1250) — compile-verified, matrix operand order checked by reading, not run on hardware
//
#include <hip/hip_runtime.h>
#include <stdint.h>

// ---------------------------------------------------------------------------
// Problem constants (match the JAX reference)
// ---------------------------------------------------------------------------
#define N_ATOMS    131072
#define NUM_ELE    89
#define NUM_GRAPH  4096
#define N_ATOM_FEAT 92
#define N_AE       128
#define EPS_BN     1e-5f

typedef __bf16 bf16;
typedef __bf16 v16bf __attribute__((ext_vector_type(16)));
typedef float  v8f   __attribute__((ext_vector_type(8)));

union ABfrag { v16bf v; uint4 q[2]; };

// ---------------------------------------------------------------------------
// Weight prep: fp32 [Kin, srcCols] (col slice of width 128 at colOff) ->
// bf16 in WMMA B-fragment layout [kchunk][ct(8)][lane(32)][j(16)].
// 16-bit B layout mirrors the A-matrix 16x32 layout from CDNA5 ISA:
//   lane<16  : col = ct*16+lane,     j<8 -> k=j,    j>=8 -> k=j+8
//   lane>=16 : col = ct*16+lane-16,  j<8 -> k=j+8,  j>=8 -> k=j+16
// k >= Kin rows zero-padded (Kpad is a multiple of 32).
// ---------------------------------------------------------------------------
__global__ void prep_w(const float* __restrict__ W, int Kin, int srcCols,
                       int colOff, bf16* __restrict__ dst, int Kpad)
{
    int idx = blockIdx.x * blockDim.x + threadIdx.x;       // Kpad*128 elements
    if (idx >= Kpad * 128) return;
    int k   = idx >> 7;
    int col = idx & 127;
    float v = (k < Kin) ? W[(size_t)k * srcCols + colOff + col] : 0.0f;
    int kc = k >> 5, kk = k & 31;
    int ct = col >> 4, cl = col & 15;
    int hi = ((kk & 15) >= 8) ? 1 : 0;
    int lane = cl + hi * 16;
    int j = (kk < 8) ? kk : (kk < 16 ? kk - 8 : (kk < 24 ? kk - 8 : kk - 16));
    dst[((size_t)((kc * 8 + ct) * 32 + lane)) * 16 + j] = (bf16)v;
}

// ---------------------------------------------------------------------------
// Tiled WMMA GEMM: Out[N,128] = act(A[N,Kin] @ W[Kin,128] + bias)
// Block = 256 threads = 8 waves; block tile = 128 rows x 128 cols.
// Each wave: 16 rows x 128 cols = 8 accumulator tiles (v8f each).
// B fragments live at constant offsets (ct*1024, +16) from a per-lane base
// pointer advancing 8192 B per k-chunk -> constant-offset global_load_b128,
// all 16 loads issued as one clause before the 8 WMMAs (load/WMMA overlap).
// A_F32:   A is fp32 (strided view, K-masked); else bf16 contiguous-K view.
// RELU:    apply relu;  OUT_F32: write fp32 pre-activation (for BN) else bf16.
// DO_STATS: per-column sum / sum-of-squares -> LDS -> 1 global atomic/col/blk.
// ---------------------------------------------------------------------------
template<int A_F32, int RELU, int OUT_F32, int DO_STATS>
__global__ __launch_bounds__(256)
void gemm128(const void* __restrict__ Ap, int lda, int acol0, int Kin, int Kpad,
             const bf16* __restrict__ Wsw, const float* __restrict__ bias,
             void* __restrict__ Out, int ldo, int ocol0,
             float* __restrict__ stats)
{
    const int tid  = threadIdx.x;
    const int wave = tid >> 5;
    const int lane = tid & 31;
    const int row0 = blockIdx.x * 128 + wave * 16;
    const int lhi  = lane >> 4;            // 0 / 1
    const int cl   = lane & 15;
    const int arow = row0 + cl;            // A-fragment row for this lane
    const int khalf = lhi * 8;

    __shared__ float ssum[128];
    __shared__ float ssq[128];
    if (DO_STATS) {
        if (tid < 128) { ssum[tid] = 0.0f; ssq[tid] = 0.0f; }
        __syncthreads();
    }

    v8f acc[8] = {};

    // Per-lane base pointers; strength-reduced strides (no per-iter addr math).
    const char* bptr = (const char*)Wsw + (size_t)lane * 32;
    const char* aptr = (const char*)Ap + ((size_t)arow * lda + khalf) * 2; // bf16 path
    const float* Af  = (const float*)Ap + (size_t)arow * lda + acol0;      // fp32 path

    const int nkc = Kpad >> 5;
    for (int kc = 0; kc < nkc; ++kc) {
        ABfrag a;
        if (A_F32) {
            const int kb = kc << 5;
            #pragma unroll
            for (int j = 0; j < 16; ++j) {
                int k = kb + khalf + j + ((j >= 8) ? 8 : 0);
                float v = (k < Kin) ? Af[k] : 0.0f;
                a.v[j] = (bf16)v;
            }
        } else {
            a.q[0] = *(const uint4*)(aptr);
            a.q[1] = *(const uint4*)(aptr + 32);   // elems +16 -> +32 bytes
            aptr += 64;                            // 32 bf16 per k-chunk
        }

        // Load all 8 B fragments first (constant offsets -> one load clause),
        // then run the 8 WMMAs back-to-back.
        ABfrag b[8];
        #pragma unroll
        for (int ct = 0; ct < 8; ++ct) {
            b[ct].q[0] = *(const uint4*)(bptr + ct * 1024);
            b[ct].q[1] = *(const uint4*)(bptr + ct * 1024 + 16);
        }
        bptr += 8192;                              // 8 tiles * 1024 B
        #pragma unroll
        for (int ct = 0; ct < 8; ++ct) {
            acc[ct] = __builtin_amdgcn_wmma_f32_16x16x32_bf16(
                false, a.v, false, b[ct].v, (short)0, acc[ct], false, false);
        }
    }

    // Epilogue. C/D layout: lane<16 -> col=lane, rows r; lane>=16 -> rows r+8.
    #pragma unroll
    for (int ct = 0; ct < 8; ++ct) {
        const int col = ct * 16 + cl;
        const float bi = bias ? bias[col] : 0.0f;
        float s = 0.0f, s2 = 0.0f;
        #pragma unroll
        for (int r = 0; r < 8; ++r) {
            int row = row0 + r + lhi * 8;
            float v = acc[ct][r] + bi;
            if (RELU) v = fmaxf(v, 0.0f);
            if (OUT_F32)
                ((float*)Out)[(size_t)row * ldo + ocol0 + col] = v;
            else
                ((bf16*)Out)[(size_t)row * ldo + ocol0 + col] = (bf16)v;
            if (DO_STATS) { s += v; s2 += v * v; }
        }
        if (DO_STATS) {
            atomicAdd(&ssum[col], s);
            atomicAdd(&ssq[col], s2);
        }
    }
    if (DO_STATS) {
        __syncthreads();
        if (tid < 128) {
            atomicAdd(&stats[tid],       ssum[tid]);
            atomicAdd(&stats[128 + tid], ssq[tid]);
        }
    }
}

// ---------------------------------------------------------------------------
// Fold BN batch stats into scale/shift: sc = g*rsqrt(var+eps), sh = be - m*sc
// ---------------------------------------------------------------------------
__global__ void finalize_bn(const float* __restrict__ stats,
                            const float* __restrict__ g,
                            const float* __restrict__ be,
                            float* __restrict__ scsh, float invN)
{
    int c = threadIdx.x;                    // 128 threads
    float m  = stats[c] * invN;
    float v  = stats[128 + c] * invN - m * m;
    float sc = g[c] * rsqrtf(v + EPS_BN);
    scsh[c]       = sc;
    scsh[128 + c] = be[c] - m * sc;
}

// ---------------------------------------------------------------------------
// h = relu(Y*sc + sh) -> bf16 out (col offset / ld selectable).
// POOL: also accumulate per-element-class sums in an LDS 89x128 table,
// flushed with one global atomic per (e,col) per block.
// ---------------------------------------------------------------------------
template<bool POOL>
__global__ __launch_bounds__(256)
void bnrelu_pool(const float* __restrict__ Y, const float* __restrict__ scsh,
                 const int* __restrict__ ele, bf16* __restrict__ outp,
                 int ldo, int ocol0,
                 float* __restrict__ poolsum, float* __restrict__ poolcnt,
                 int rowsPerBlock)
{
    __shared__ float ps[POOL ? NUM_ELE * 128 : 1];
    __shared__ float pc[POOL ? NUM_ELE : 1];
    const int tid  = threadIdx.x;
    const int col  = tid & 127;
    const int half = tid >> 7;              // two row-streams per column
    if (POOL) {
        for (int i = tid; i < NUM_ELE * 128; i += 256) ps[i] = 0.0f;
        for (int i = tid; i < NUM_ELE; i += 256) pc[i] = 0.0f;
        __syncthreads();
    }
    const float sc = scsh[col];
    const float sh = scsh[128 + col];
    const int rbase = blockIdx.x * rowsPerBlock;
    for (int i = 0; i < rowsPerBlock; i += 2) {
        int row = rbase + i + half;
        float h = fmaxf(Y[(size_t)row * 128 + col] * sc + sh, 0.0f);
        outp[(size_t)row * ldo + ocol0 + col] = (bf16)h;
        if (POOL) {
            int e = ele[row];
            atomicAdd(&ps[e * 128 + col], h);
            if (col == 0) atomicAdd(&pc[e], 1.0f);
        }
    }
    if (POOL) {
        __syncthreads();
        for (int i = tid; i < NUM_ELE * 128; i += 256) atomicAdd(&poolsum[i], ps[i]);
        for (int i = tid; i < NUM_ELE; i += 256) atomicAdd(&poolcnt[i], pc[i]);
    }
}

// pooled[ele_idx[row]] -> xcat cols 128..255 (ld 512)
__global__ void pool_gather(const float* __restrict__ poolsum,
                            const float* __restrict__ poolcnt,
                            const int* __restrict__ ele,
                            bf16* __restrict__ xcat)
{
    int idx = blockIdx.x * blockDim.x + threadIdx.x;   // N*128
    int row = idx >> 7, col = idx & 127;
    int e = ele[row];
    float v = poolsum[e * 128 + col] / fmaxf(poolcnt[e], 1.0f);
    xcat[(size_t)row * 512 + 128 + col] = (bf16)v;
}

// copy x (bf16 [N,256]) into xcat cols 0..255 (as packed u32)
__global__ void copy_x(const uint32_t* __restrict__ xb, uint32_t* __restrict__ xc)
{
    int idx = blockIdx.x * blockDim.x + threadIdx.x;   // N*128 u32
    int row = idx >> 7, c = idx & 127;
    xc[(size_t)row * 256 + c] = xb[idx];
}

// Graph pooling: graph_idx is sorted -> register run-accumulation, flush on
// segment change (few atomics per block instead of per row).
__global__ __launch_bounds__(256)
void graph_pool(const bf16* __restrict__ h, const int* __restrict__ gidx,
                float* __restrict__ gsum, float* __restrict__ gcnt,
                int rowsPerBlock)
{
    int col = threadIdx.x;                 // 0..255
    int rbase = blockIdx.x * rowsPerBlock;
    int curg = gidx[rbase];
    float acc = 0.0f, cnt = 0.0f;
    for (int i = 0; i < rowsPerBlock; ++i) {
        int row = rbase + i;
        int g = gidx[row];
        if (g != curg) {
            atomicAdd(&gsum[(size_t)curg * 256 + col], acc);
            if (col == 0) atomicAdd(&gcnt[curg], cnt);
            curg = g; acc = 0.0f; cnt = 0.0f;
        }
        acc += (float)h[(size_t)row * 256 + col];
        cnt += 1.0f;
    }
    atomicAdd(&gsum[(size_t)curg * 256 + col], acc);
    if (col == 0) atomicAdd(&gcnt[curg], cnt);
}

// out[g] = dot(gsum[g]/max(cnt,1), w) + b
__global__ __launch_bounds__(256)
void final_fc(const float* __restrict__ gsum, const float* __restrict__ gcnt,
              const float* __restrict__ w, const float* __restrict__ b,
              float* __restrict__ out)
{
    __shared__ float red[256];
    int g = blockIdx.x, t = threadIdx.x;
    float inv = 1.0f / fmaxf(gcnt[g], 1.0f);
    red[t] = gsum[(size_t)g * 256 + t] * inv * w[t];
    __syncthreads();
    for (int s = 128; s > 0; s >>= 1) {
        if (t < s) red[t] += red[t + s];
        __syncthreads();
    }
    if (t == 0) out[g] = red[0] + b[0];
}

// ---------------------------------------------------------------------------
// Host-side orchestration.
// Input order: setup_inputs() dict order; `params` pytree flattened with
// dict keys sorted alphabetically at every level (JAX tree flattening):
//  0 atom_feat  1 rdf_feat  2 bdf_feat  3 atom_idx  4 ele_idx  5 graph_idx
//  6 ref_feat
//  bdf_layer: 7 bn1.be 8 bn1.g 9 fc1.b 10 fc1.w 11 l1.bn.be 12 l1.bn.g
//             13 l1.fc.b 14 l1.fc.w 15 l2.bn.be 16 l2.bn.g 17 l2.fc.b 18 l2.fc.w
//  19 embed_atom.b 20 embed_atom.w 21 embed_bdf.b 22 embed_bdf.w
//  23 embed_rdf.b 24 embed_rdf.w 25 fc1.b 26 fc1.w
//  rdf_layer: 27..38 (same order as bdf_layer)
// ---------------------------------------------------------------------------
extern "C" void kernel_launch(void* const* d_in, const int* in_sizes, int n_in,
                              void* d_out, int out_size, void* d_ws, size_t ws_size,
                              hipStream_t stream)
{
    (void)in_sizes; (void)n_in; (void)out_size; (void)ws_size;
    const float* atom_feat = (const float*)d_in[0];
    const float* rdf_feat  = (const float*)d_in[1];
    const float* bdf_feat  = (const float*)d_in[2];
    const int*   ele_idx   = (const int*)d_in[4];
    const int*   graph_idx = (const int*)d_in[5];

    char* ws = (char*)d_ws;
    size_t off = 0;
    auto alloc = [&](size_t bytes) -> char* {
        char* p = ws + off;
        off = (off + bytes + 255) & ~(size_t)255;
        return p;
    };

    bf16*  xbuf   = (bf16*)alloc((size_t)N_ATOMS * 256 * 2);  // atom embeds
    bf16*  xcat   = (bf16*)alloc((size_t)N_ATOMS * 512 * 2);  // layer inputs
    float* preY   = (float*)alloc((size_t)N_ATOMS * 128 * 4); // pre-activation
    bf16*  hfinal = (bf16*)alloc((size_t)N_ATOMS * 256 * 2);  // [h1|h2]
    float* stats  = (float*)alloc(256 * 4);
    float* scsh   = (float*)alloc(256 * 4);
    float* poolsum = (float*)alloc(NUM_ELE * 128 * 4);
    float* poolcnt = (float*)alloc(NUM_ELE * 4);
    float* gsum   = (float*)alloc((size_t)NUM_GRAPH * 256 * 4);
    float* gcnt   = (float*)alloc(NUM_GRAPH * 4);
    bf16* wAtom = (bf16*)alloc(96 * 128 * 2);
    bf16* wD0[2] = { (bf16*)alloc(128 * 128 * 2), (bf16*)alloc(128 * 128 * 2) };
    bf16* wD1[2] = { (bf16*)alloc(128 * 128 * 2), (bf16*)alloc(128 * 128 * 2) };
    bf16* wL1[2] = { (bf16*)alloc(512 * 128 * 2), (bf16*)alloc(512 * 128 * 2) };
    bf16* wL2[2] = { (bf16*)alloc(512 * 128 * 2), (bf16*)alloc(512 * 128 * 2) };
    bf16* wFc[2] = { (bf16*)alloc(256 * 128 * 2), (bf16*)alloc(256 * 128 * 2) };

    const int GEMM_GRID = N_ATOMS / 128;                // 1024
    const int ELEM_GRID = (N_ATOMS * 128) / 256;        // 65536
    auto pg = [](int elems) { return (elems + 255) / 256; };

    // ---- weight prep (bf16 + B-fragment swizzle) ----
    prep_w<<<pg(96 * 128), 256, 0, stream>>>((const float*)d_in[20], 92, 128, 0, wAtom, 96);
    prep_w<<<pg(128 * 128), 256, 0, stream>>>((const float*)d_in[24], 128, 256, 0,   wD0[0], 128);
    prep_w<<<pg(128 * 128), 256, 0, stream>>>((const float*)d_in[24], 128, 256, 128, wD1[0], 128);
    prep_w<<<pg(128 * 128), 256, 0, stream>>>((const float*)d_in[22], 128, 256, 0,   wD0[1], 128);
    prep_w<<<pg(128 * 128), 256, 0, stream>>>((const float*)d_in[22], 128, 256, 128, wD1[1], 128);
    prep_w<<<pg(512 * 128), 256, 0, stream>>>((const float*)d_in[34], 512, 128, 0, wL1[0], 512);
    prep_w<<<pg(512 * 128), 256, 0, stream>>>((const float*)d_in[38], 512, 128, 0, wL2[0], 512);
    prep_w<<<pg(256 * 128), 256, 0, stream>>>((const float*)d_in[30], 256, 128, 0, wFc[0], 256);
    prep_w<<<pg(512 * 128), 256, 0, stream>>>((const float*)d_in[14], 512, 128, 0, wL1[1], 512);
    prep_w<<<pg(512 * 128), 256, 0, stream>>>((const float*)d_in[18], 512, 128, 0, wL2[1], 512);
    prep_w<<<pg(256 * 128), 256, 0, stream>>>((const float*)d_in[10], 256, 128, 0, wFc[1], 256);

    // ---- atom embeddings: x = [relu(Wa) | relu(Wa on 2nd half)] ----
    const float* embA_b = (const float*)d_in[19];
    gemm128<1, 1, 0, 0><<<GEMM_GRID, 256, 0, stream>>>(
        atom_feat, 2 * N_ATOM_FEAT, 0, N_ATOM_FEAT, 96, wAtom, embA_b,
        xbuf, 256, 0, nullptr);
    gemm128<1, 1, 0, 0><<<GEMM_GRID, 256, 0, stream>>>(
        atom_feat, 2 * N_ATOM_FEAT, N_ATOM_FEAT, N_ATOM_FEAT, 96, wAtom, embA_b,
        xbuf, 256, 128, nullptr);

    const float invN = 1.0f / (float)N_ATOMS;

    for (int m = 0; m < 2; ++m) {          // m=0: rdf module, m=1: bdf module
        const float* feat   = (m == 0) ? rdf_feat : bdf_feat;
        const float* embD_b = (const float*)((m == 0) ? d_in[23] : d_in[21]);
        const int bi = (m == 0) ? 27 : 7;  // layer param base index
        const float* bn1be = (const float*)d_in[bi + 0];
        const float* bn1g  = (const float*)d_in[bi + 1];
        const float* fcb   = (const float*)d_in[bi + 2];
        const float* l1be  = (const float*)d_in[bi + 4];
        const float* l1g   = (const float*)d_in[bi + 5];
        const float* l1b   = (const float*)d_in[bi + 6];
        const float* l2be  = (const float*)d_in[bi + 8];
        const float* l2g   = (const float*)d_in[bi + 9];
        const float* l2b   = (const float*)d_in[bi + 10];

        // dist embed -> xcat cols 256..511 ; x -> cols 0..255
        gemm128<1, 1, 0, 0><<<GEMM_GRID, 256, 0, stream>>>(
            feat, 128, 0, 128, 128, wD0[m], embD_b, xcat, 512, 256, nullptr);
        gemm128<1, 1, 0, 0><<<GEMM_GRID, 256, 0, stream>>>(
            feat, 128, 0, 128, 128, wD1[m], embD_b + 128, xcat, 512, 384, nullptr);
        copy_x<<<ELEM_GRID, 256, 0, stream>>>((const uint32_t*)xbuf, (uint32_t*)xcat);

        // ---- l1 ----
        hipMemsetAsync(stats, 0, 256 * 4, stream);
        gemm128<0, 0, 1, 1><<<GEMM_GRID, 256, 0, stream>>>(
            xcat, 512, 0, 512, 512, wL1[m], l1b, preY, 128, 0, stats);
        finalize_bn<<<1, 128, 0, stream>>>(stats, l1g, l1be, scsh, invN);
        hipMemsetAsync(poolsum, 0, NUM_ELE * 128 * 4, stream);
        hipMemsetAsync(poolcnt, 0, NUM_ELE * 4, stream);
        bnrelu_pool<true><<<N_ATOMS / 512, 256, 0, stream>>>(
            preY, scsh, ele_idx, xcat, 512, 0, poolsum, poolcnt, 512);
        pool_gather<<<ELEM_GRID, 256, 0, stream>>>(poolsum, poolcnt, ele_idx, xcat);

        // ---- l2 ----
        hipMemsetAsync(stats, 0, 256 * 4, stream);
        gemm128<0, 0, 1, 1><<<GEMM_GRID, 256, 0, stream>>>(
            xcat, 512, 0, 512, 512, wL2[m], l2b, preY, 128, 0, stats);
        finalize_bn<<<1, 128, 0, stream>>>(stats, l2g, l2be, scsh, invN);
        hipMemsetAsync(poolsum, 0, NUM_ELE * 128 * 4, stream);
        hipMemsetAsync(poolcnt, 0, NUM_ELE * 4, stream);
        bnrelu_pool<true><<<N_ATOMS / 512, 256, 0, stream>>>(
            preY, scsh, ele_idx, xcat, 512, 0, poolsum, poolcnt, 512);
        pool_gather<<<ELEM_GRID, 256, 0, stream>>>(poolsum, poolcnt, ele_idx, xcat);

        // ---- fc1 (Kin=256 view of xcat) ----
        hipMemsetAsync(stats, 0, 256 * 4, stream);
        gemm128<0, 0, 1, 1><<<GEMM_GRID, 256, 0, stream>>>(
            xcat, 512, 0, 256, 256, wFc[m], fcb, preY, 128, 0, stats);
        finalize_bn<<<1, 128, 0, stream>>>(stats, bn1g, bn1be, scsh, invN);
        bnrelu_pool<false><<<N_ATOMS / 512, 256, 0, stream>>>(
            preY, scsh, nullptr, hfinal, 256, m * 128, nullptr, nullptr, 512);
    }

    // ---- graph pooling + final projection ----
    hipMemsetAsync(gsum, 0, (size_t)NUM_GRAPH * 256 * 4, stream);
    hipMemsetAsync(gcnt, 0, NUM_GRAPH * 4, stream);
    graph_pool<<<N_ATOMS / 128, 256, 0, stream>>>(hfinal, graph_idx, gsum, gcnt, 128);
    final_fc<<<NUM_GRAPH, 256, 0, stream>>>(
        gsum, gcnt, (const float*)d_in[26], (const float*)d_in[25], (float*)d_out);
}
